// cheby_conv_ds_15513421873578
// MI455X (gfx1250) — matrix operand order repaired and checked by reading, not compiled, not run on hardware
//
#include <hip/hip_runtime.h>

// Chebyshev graph conv (K=3) for MI455X / gfx1250.
// Big GEMMs use split-precision "bf16x3" on V_WMMA_F32_16X16X32_BF16:
//   f32 operand v = hi + lo (both bf16);  A*B ~= Ah*Bh + Ah*Bl + Al*Bh  (f32 accum)
// -> ~16 mantissa bits/operand (~1e-5 rel err), ~5x the f32-WMMA pipe throughput.
//
// Math:  out[b,o,q,l] = bias[o]
//                     + ds[b,q,q] * (W0 @ x)[b,o,q,l]                        (k=0, diagonal)
//                     + (W1 @ ((adj .* ds_b) @ x))[b,o,q,l]                  (k=1)
//                     + (W2 @ ((T2  .* ds_b) @ x))[b,o,q,l],  T2 = 2*adj^2-I (k=2)
// where Wk[o,c] = W[o, c*3+k].
//
// Workspace: [ T2 : 512*512 f32 (1 MB) ][ Y : 32*64*512*64 f32 (256 MB) ]

typedef __bf16 bf16_t;
typedef __bf16 v2bf  __attribute__((ext_vector_type(2)));
typedef __bf16 v8bf  __attribute__((ext_vector_type(8)));
typedef __bf16 v16bf __attribute__((ext_vector_type(16)));
typedef float  v2f   __attribute__((ext_vector_type(2)));
typedef float  v8f   __attribute__((ext_vector_type(8)));

#define B_   32
#define C_   64          // c_in == c_out
#define N_   512
#define L_   64
#define KCH  3
#define NL_  (N_ * L_)   // 32768
#define CL_  (C_ * L_)   // 4096
#define LSTR 40          // LDS row stride in bf16 elems (80 B: 16B-aligned, low conflict)

static __device__ __forceinline__ v8f wmma_bf16(v16bf a, v16bf b, v8f c) {
  return __builtin_amdgcn_wmma_f32_16x16x32_bf16(
      /*neg_a=*/false, a, /*neg_b=*/false, b,
      /*c_mod=*/(short)0, c, /*reuse_a=*/false, /*reuse_b=*/false);
}

// A-fragment (16x32 bf16): lane t -> K chunks [8*(t>>4),+8) and [8*(t>>4)+16,+8)
static __device__ __forceinline__ v16bf frag_a(const bf16_t* row, int kb8) {
  v8bf c0 = *(const v8bf*)(row + kb8);
  v8bf c1 = *(const v8bf*)(row + kb8 + 16);
  return __builtin_shufflevector(c0, c1, 0, 1, 2, 3, 4, 5, 6, 7,
                                         8, 9, 10, 11, 12, 13, 14, 15);
}
// B-fragment (32x16 bf16): lane t -> K chunks [8*(t>>4),+8) and [8*(t>>4)+16,+8)
static __device__ __forceinline__ v16bf frag_b(const bf16_t* col, int kb16) {
  v8bf c0 = *(const v8bf*)(col + (kb16 >> 1));
  v8bf c1 = *(const v8bf*)(col + (kb16 >> 1) + 16);
  return __builtin_shufflevector(c0, c1, 0, 1, 2, 3, 4, 5, 6, 7,
                                         8, 9, 10, 11, 12, 13, 14, 15);
}

static __device__ __forceinline__ void split2(float v0, float v1, v2bf* hi, v2bf* lo) {
  bf16_t h0 = (bf16_t)v0, h1 = (bf16_t)v1;
  bf16_t l0 = (bf16_t)(v0 - (float)h0), l1 = (bf16_t)(v1 - (float)h1);
  v2bf h = {h0, h1}, l = {l0, l1};
  *hi = h; *lo = l;
}

// ---------------------------------------------------------------------------
// Kernel 1: T2 = 2 * adj @ adj - I   (512x512, exact f32, negligible cost)
// ---------------------------------------------------------------------------
__global__ __launch_bounds__(256) void t2_kernel(const float* __restrict__ adj,
                                                 float* __restrict__ t2) {
  __shared__ float As[16][17];
  __shared__ float Bs[16][17];
  const int tx = threadIdx.x & 15;
  const int ty = threadIdx.x >> 4;
  const int row = blockIdx.y * 16 + ty;
  const int col = blockIdx.x * 16 + tx;
  float s = 0.f;
  for (int kb = 0; kb < N_; kb += 16) {
    As[ty][tx] = adj[row * N_ + kb + tx];
    Bs[ty][tx] = adj[(kb + ty) * N_ + col];
    __syncthreads();
#pragma unroll
    for (int k = 0; k < 16; ++k) s += As[ty][k] * Bs[k][tx];
    __syncthreads();
  }
  t2[row * N_ + col] = 2.f * s - (row == col ? 1.f : 0.f);
}

// ---------------------------------------------------------------------------
// Kernel 2: node GEMM  Y[b,c,q,l] = sum_n (T[q,n]*ds[b,q,n]) * x[b,c,n,l]
// Block tile 128(q) x 128(c,l), K-block 32 (one bf16-WMMA K), 8 waves,
// each wave 32x64 via 2x4 grid of 16x16 accumulators, 3 WMMA per tile/K-block.
// ---------------------------------------------------------------------------
__global__ __launch_bounds__(256) void cheby_node_gemm(
    const float* __restrict__ x,   // [B, C, N, L]
    const float* __restrict__ ds,  // [B, N, N]
    const float* __restrict__ T,   // [N, N]  (adj or T2)
    float* __restrict__ Y) {       // [B, C, N, L]
  __shared__ __align__(16) bf16_t Ah[128 * LSTR];  // [q-row][k] hi
  __shared__ __align__(16) bf16_t Al[128 * LSTR];  // lo
  __shared__ __align__(16) bf16_t Bh[128 * LSTR];  // [col][k] (transposed x) hi
  __shared__ __align__(16) bf16_t Bl[128 * LSTR];  // lo

  const int b       = blockIdx.z;
  const int qBase   = blockIdx.y * 128;
  const int colBase = blockIdx.x * 128;
  const float* dsb = ds + (size_t)b * N_ * N_;
  const float* xb  = x  + (size_t)b * C_ * NL_;

  const int t    = threadIdx.x;
  const int lane = t & 31;
  const int wid  = t >> 5;
  const int wm   = (wid & 3) * 32;   // 4 waves along q
  const int wn   = (wid >> 2) * 64;  // 2 waves along cols
  const int lm   = lane & 15;
  const int kb8  = (lane >> 4) * 8;  // A-frag K sub-offset
  const int kb16 = (lane >> 4) * 16; // B-frag K sub-offset

  const v8f vzero = {0.f, 0.f, 0.f, 0.f, 0.f, 0.f, 0.f, 0.f};
  v8f acc[2][4];
  for (int mi = 0; mi < 2; ++mi)
    for (int ni = 0; ni < 4; ++ni) acc[mi][ni] = vzero;

  for (int nb = 0; nb < N_; nb += 32) {
    // A tile: 128 q-rows x 32 n of (T .* ds_b); 2 consecutive n per thread
#pragma unroll
    for (int i = 0; i < 8; ++i) {
      int e = t + i * 256;               // 2048 pairs
      int r = e >> 4, c0 = (e & 15) * 2;
      int q = qBase + r, n0 = nb + c0;
      v2f tv = *(const v2f*)&T[q * N_ + n0];
      v2f dv = *(const v2f*)&dsb[q * N_ + n0];
      v2bf h, l;
      split2(tv.x * dv.x, tv.y * dv.y, &h, &l);
      *(v2bf*)&Ah[r * LSTR + c0] = h;
      *(v2bf*)&Al[r * LSTR + c0] = l;
    }
    // B tile: 32 n-rows x 128 (c,l)-cols of x, stored transposed; 2 n per thread
#pragma unroll
    for (int i = 0; i < 8; ++i) {
      int e = t + i * 256;               // 2048 pairs
      int j = e & 127, r0 = (e >> 7) * 2;
      int gcol = colBase + j;
      int c = gcol >> 6, l = gcol & 63;
      const float* p = xb + (size_t)c * NL_ + (nb + r0) * L_ + l;
      v2bf h, lo;
      split2(p[0], p[L_], &h, &lo);
      *(v2bf*)&Bh[j * LSTR + r0] = h;
      *(v2bf*)&Bl[j * LSTR + r0] = lo;
    }
    __syncthreads();

    v16bf a0h = frag_a(&Ah[(wm + lm) * LSTR], kb8);
    v16bf a0l = frag_a(&Al[(wm + lm) * LSTR], kb8);
    v16bf a1h = frag_a(&Ah[(wm + 16 + lm) * LSTR], kb8);
    v16bf a1l = frag_a(&Al[(wm + 16 + lm) * LSTR], kb8);
#pragma unroll
    for (int ni = 0; ni < 4; ++ni) {
      v16bf bnh = frag_b(&Bh[(wn + ni * 16 + lm) * LSTR], kb16);
      v16bf bnl = frag_b(&Bl[(wn + ni * 16 + lm) * LSTR], kb16);
      acc[0][ni] = wmma_bf16(a0h, bnh, acc[0][ni]);
      acc[0][ni] = wmma_bf16(a0l, bnh, acc[0][ni]);
      acc[0][ni] = wmma_bf16(a0h, bnl, acc[0][ni]);
      acc[1][ni] = wmma_bf16(a1h, bnh, acc[1][ni]);
      acc[1][ni] = wmma_bf16(a1l, bnh, acc[1][ni]);
      acc[1][ni] = wmma_bf16(a1h, bnl, acc[1][ni]);
    }
    __syncthreads();
  }

  // Epilogue: C/D layout -> row = 8*(lane>>4)+r, col = lane&15
  float* Yb = Y + (size_t)b * C_ * NL_;
#pragma unroll
  for (int mi = 0; mi < 2; ++mi) {
#pragma unroll
    for (int ni = 0; ni < 4; ++ni) {
      int gcol = colBase + wn + ni * 16 + lm;
      int c = gcol >> 6, l = gcol & 63;
      int q0 = qBase + wm + mi * 16 + (lane >> 4) * 8;
      float* dst = Yb + (size_t)c * NL_ + q0 * L_ + l;
#pragma unroll
      for (int r = 0; r < 8; ++r) dst[r * L_] = acc[mi][ni][r];
    }
  }
}

// ---------------------------------------------------------------------------
// Kernel 3: channel-mix GEMM  (M=64, K=64, 32768 cols per batch), bf16x3
//   initMode=1: out = ds[b,q,q] * (Wk @ S) + bias      (k=0, S=x)
//   initMode=0: out += Wk @ S                          (k=1,2, S=Y)
// ---------------------------------------------------------------------------
__global__ __launch_bounds__(256) void mix_gemm(
    const float* __restrict__ S,     // [B, C, N, L]
    const float* __restrict__ W,     // [64, 192]
    const float* __restrict__ bias,  // [64]
    const float* __restrict__ ds,    // [B, N, N]
    float* __restrict__ out,         // [B, 64, N, L]
    int kSel, int initMode) {
  __shared__ __align__(16) bf16_t Wh[64 * LSTR];
  __shared__ __align__(16) bf16_t Wl[64 * LSTR];
  __shared__ __align__(16) bf16_t Bh[256 * LSTR];
  __shared__ __align__(16) bf16_t Bl[256 * LSTR];

  const int b       = blockIdx.z;
  const int colBase = blockIdx.x * 256;
  const float* Sb = S + (size_t)b * C_ * NL_;

  const int t    = threadIdx.x;
  const int lane = t & 31;
  const int wid  = t >> 5;
  const int wm   = (wid & 1) * 32;
  const int wn   = (wid >> 1) * 64;
  const int lm   = lane & 15;
  const int kb8  = (lane >> 4) * 8;
  const int kb16 = (lane >> 4) * 16;

  const v8f vzero = {0.f, 0.f, 0.f, 0.f, 0.f, 0.f, 0.f, 0.f};
  v8f acc[2][4];
  for (int mi = 0; mi < 2; ++mi)
    for (int ni = 0; ni < 4; ++ni) acc[mi][ni] = vzero;

  for (int kb = 0; kb < C_; kb += 32) {
    // Wk tile: 64 x 32; 2 consecutive k (c) per thread
#pragma unroll
    for (int i = 0; i < 4; ++i) {
      int e = t + i * 256;               // 1024 pairs
      int o = e >> 4, k0 = (e & 15) * 2;
      const float* p = W + o * (C_ * KCH) + (kb + k0) * KCH + kSel;
      v2bf h, l;
      split2(p[0], p[KCH], &h, &l);
      *(v2bf*)&Wh[o * LSTR + k0] = h;
      *(v2bf*)&Wl[o * LSTR + k0] = l;
    }
    // S tile: 32 c-rows x 256 (q,l)-cols, (q,l) contiguous -> coalesced
#pragma unroll
    for (int i = 0; i < 16; ++i) {
      int e = t + i * 256;               // 4096 pairs
      int j = e & 255, r0 = (e >> 8) * 2;
      const float* p = Sb + (size_t)(kb + r0) * NL_ + colBase + j;
      v2bf h, l;
      split2(p[0], p[NL_], &h, &l);
      *(v2bf*)&Bh[j * LSTR + r0] = h;
      *(v2bf*)&Bl[j * LSTR + r0] = l;
    }
    __syncthreads();

    v16bf a0h = frag_a(&Wh[(wm + lm) * LSTR], kb8);
    v16bf a0l = frag_a(&Wl[(wm + lm) * LSTR], kb8);
    v16bf a1h = frag_a(&Wh[(wm + 16 + lm) * LSTR], kb8);
    v16bf a1l = frag_a(&Wl[(wm + 16 + lm) * LSTR], kb8);
#pragma unroll
    for (int ni = 0; ni < 4; ++ni) {
      v16bf bnh = frag_b(&Bh[(wn + ni * 16 + lm) * LSTR], kb16);
      v16bf bnl = frag_b(&Bl[(wn + ni * 16 + lm) * LSTR], kb16);
      acc[0][ni] = wmma_bf16(a0h, bnh, acc[0][ni]);
      acc[0][ni] = wmma_bf16(a0l, bnh, acc[0][ni]);
      acc[0][ni] = wmma_bf16(a0h, bnl, acc[0][ni]);
      acc[1][ni] = wmma_bf16(a1h, bnh, acc[1][ni]);
      acc[1][ni] = wmma_bf16(a1l, bnh, acc[1][ni]);
      acc[1][ni] = wmma_bf16(a1h, bnl, acc[1][ni]);
    }
    __syncthreads();
  }

  float* outb = out + (size_t)b * C_ * NL_;
  const float* dsb = ds + (size_t)b * N_ * N_;
#pragma unroll
  for (int mi = 0; mi < 2; ++mi) {
#pragma unroll
    for (int ni = 0; ni < 4; ++ni) {
      int gcol = colBase + wn + ni * 16 + lm;
      int q  = gcol >> 6;
      int o0 = wm + mi * 16 + (lane >> 4) * 8;
      float* dst = outb + (size_t)o0 * NL_ + gcol;
      if (initMode) {
        float dsd = dsb[q * N_ + q];
#pragma unroll
        for (int r = 0; r < 8; ++r)
          dst[(size_t)r * NL_] = acc[mi][ni][r] * dsd + bias[o0 + r];
      } else {
#pragma unroll
        for (int r = 0; r < 8; ++r)
          dst[(size_t)r * NL_] += acc[mi][ni][r];
      }
    }
  }
}

// ---------------------------------------------------------------------------
extern "C" void kernel_launch(void* const* d_in, const int* in_sizes, int n_in,
                              void* d_out, int out_size, void* d_ws, size_t ws_size,
                              hipStream_t stream) {
  (void)in_sizes; (void)n_in; (void)out_size; (void)ws_size;
  const float* x    = (const float*)d_in[0];  // [32,64,512,64]
  const float* adj  = (const float*)d_in[1];  // [512,512]
  const float* ds   = (const float*)d_in[2];  // [32,512,512]
  const float* W    = (const float*)d_in[3];  // [64,192]
  const float* bias = (const float*)d_in[4];  // [64]
  float* out = (float*)d_out;                 // [32,64,512,64]

  float* t2 = (float*)d_ws;                   // 512*512 f32
  float* Y  = t2 + (size_t)N_ * N_;           // 32*64*512*64 f32

  // T2 = 2*adj@adj - I
  t2_kernel<<<dim3(N_ / 16, N_ / 16, 1), 256, 0, stream>>>(adj, t2);
  // out = bias + ds_diag * (W0 @ x)
  mix_gemm<<<dim3(NL_ / 256, 1, B_), 256, 0, stream>>>(x, W, bias, ds, out, 0, 1);
  // Y = (adj .* ds_b) @ x ; out += W1 @ Y
  cheby_node_gemm<<<dim3(CL_ / 128, N_ / 128, B_), 256, 0, stream>>>(x, ds, adj, Y);
  mix_gemm<<<dim3(NL_ / 256, 1, B_), 256, 0, stream>>>(Y, W, bias, ds, out, 1, 0);
  // Y = (T2 .* ds_b) @ x ; out += W2 @ Y
  cheby_node_gemm<<<dim3(CL_ / 128, N_ / 128, B_), 256, 0, stream>>>(x, ds, t2, Y);
  mix_gemm<<<dim3(NL_ / 256, 1, B_), 256, 0, stream>>>(Y, W, bias, ds, out, 2, 0);
}
